// LSTMRepresentationModel_74062416053113
// MI455X (gfx1250) — matrix operand order repaired and checked by reading, not compiled, not run on hardware
//
#include <hip/hip_runtime.h>
#include <math.h>

typedef __attribute__((ext_vector_type(16))) _Float16 v16h;
typedef __attribute__((ext_vector_type(8)))  _Float16 v8h;
typedef __attribute__((ext_vector_type(8)))  float    v8f;
typedef __attribute__((ext_vector_type(4)))  _Float16 v4h;

#define HH 64   // hidden size

union Frag16 {
  v16h h16;
  _Float16 h[16];
  unsigned u[8];
};

// Hardware-native activations: one v_exp_f32 + one v_rcp_f32, branch-free.
__device__ __forceinline__ float fast_tanh(float x) {
  float e = __builtin_amdgcn_exp2f(-2.885390082f * x);   // exp(-2x)
  return 2.0f * __builtin_amdgcn_rcpf(1.0f + e) - 1.0f;
}

// ---------------------------------------------------------------------------
// f32 -> f16 conversion, 4 elements/thread
// ---------------------------------------------------------------------------
__global__ void cvt_f32_f16(const float* __restrict__ in, _Float16* __restrict__ out, int n4) {
  int i = blockIdx.x * blockDim.x + threadIdx.x;
  if (i < n4) {
    const float4 v = reinterpret_cast<const float4*>(in)[i];
    v4h o;
    o.x = (_Float16)v.x; o.y = (_Float16)v.y;
    o.z = (_Float16)v.z; o.w = (_Float16)v.w;
    reinterpret_cast<v4h*>(out)[i] = o;
  }
}

// ---------------------------------------------------------------------------
// Fused LSTM layer. Grid = B/16 workgroups, 128 threads (4 waves).
// Wave w handles gate w (64 gate columns = 4 WMMA N-tiles).
// Weights live in VGPR fragments across all T steps; x_t fragments are
// double-buffered (software pipelined two steps ahead); h/c recurrence in
// LDS/registers. IK = I/32 (1 for layer1 input=32, 2 for layer2 input=64).
// ---------------------------------------------------------------------------
template<int IK>
__global__ void __launch_bounds__(128)
lstm_layer_kernel(const _Float16* __restrict__ xin,  // [B,T,I] f16
                  const float* __restrict__ wih,     // [256, I]
                  const float* __restrict__ whh,     // [256, 64]
                  const float* __restrict__ bih,     // [256]
                  const float* __restrict__ bhh,     // [256]
                  _Float16* __restrict__ seq_out,    // [B,T,64] f16 or null
                  float* __restrict__ hT_out,        // [B,64] f32 or null
                  int T) {
  constexpr int I = IK * 32;
  __shared__ float    gbuf[4][16][HH];  // activated gates i,f,g,o
  __shared__ _Float16 hbuf[16][HH];     // hidden state, f16, row-major

  const int tid    = threadIdx.x;
  const int wv     = tid >> 5;          // wave id 0..3 == gate id
  const int lane   = tid & 31;
  const int lrow   = lane & 15;         // M (A/C) or N (B) within a 16-tile
  const int khalfA = (lane >> 4) * 8;   // A-layout K base for halves 0..7
  const int kbaseB = (lane >> 4) * 16;  // contiguous K base, unused (B loads use khalfA)
  const int b0     = blockIdx.x * 16;

  // ---- persistent weight fragments (B operands), loaded once ----
  Frag16 fwih[4][IK];
  Frag16 fwhh[4][2];
#pragma unroll
  for (int nt = 0; nt < 4; ++nt) {
    const int n = wv * 64 + nt * 16 + lrow;  // gate-vector column (0..255)
#pragma unroll
    for (int kc = 0; kc < IK; ++kc) {
      const float* row = wih + (size_t)n * I + kc * 32 + khalfA;
#pragma unroll
      for (int j = 0; j < 16; ++j) fwih[nt][kc].h[j] = (_Float16)row[j + (j & 8)];
    }
#pragma unroll
    for (int kc = 0; kc < 2; ++kc) {
      const float* row = whh + (size_t)n * HH + kc * 32 + khalfA;
#pragma unroll
      for (int j = 0; j < 16; ++j) fwhh[nt][kc].h[j] = (_Float16)row[j + (j & 8)];
    }
  }
  float bias[4];
#pragma unroll
  for (int nt = 0; nt < 4; ++nt) {
    const int n = wv * 64 + nt * 16 + lrow;
    bias[nt] = bih[n] + bhh[n];
  }

  // zero hidden state
  for (int i = tid; i < 16 * HH; i += 128) ((_Float16*)hbuf)[i] = (_Float16)0.f;

  // per-thread cell state: row um = tid>>3, cols un0..un0+7, VGPR-resident
  float cst[8];
#pragma unroll
  for (int k = 0; k < 8; ++k) cst[k] = 0.f;
  const int  um  = tid >> 3;
  const int  un0 = (tid & 7) * 8;
  const bool isg = (wv == 2);

  const _Float16* xrow = xin + (size_t)(b0 + lrow) * T * I;

  auto loadA = [&](Frag16 (&ax)[IK], int t) {
    const _Float16* xs = xrow + (size_t)t * I;
#pragma unroll
    for (int kc = 0; kc < IK; ++kc) {
      const _Float16* s = xs + kc * 32;
#pragma unroll
      for (int u = 0; u < 4; ++u)
        ax[kc].u[u] = *reinterpret_cast<const unsigned*>(s + khalfA + 2 * u);
#pragma unroll
      for (int u = 0; u < 4; ++u)
        ax[kc].u[4 + u] = *reinterpret_cast<const unsigned*>(s + 16 + khalfA + 2 * u);
    }
  };

  Frag16 axA[IK], axB[IK];
  loadA(axA, 0);
  loadA(axB, 1);

  __syncthreads();

  auto step = [&](int t, Frag16 (&ax)[IK]) {
    // ---- hidden-state A fragments from LDS ----
    Frag16 ah[2];
#pragma unroll
    for (int kc = 0; kc < 2; ++kc) {
      const _Float16* s = &hbuf[lrow][kc * 32];
#pragma unroll
      for (int u = 0; u < 4; ++u)
        ah[kc].u[u] = *reinterpret_cast<const unsigned*>(s + khalfA + 2 * u);
#pragma unroll
      for (int u = 0; u < 4; ++u)
        ah[kc].u[4 + u] = *reinterpret_cast<const unsigned*>(s + 16 + khalfA + 2 * u);
    }

    // ---- gates = bias + x_t @ Wih^T + h @ Whh^T (WMMA, f32 accum) ----
    v8f accs[4];
#pragma unroll
    for (int nt = 0; nt < 4; ++nt) {
      v8f acc;
#pragma unroll
      for (int k = 0; k < 8; ++k) acc[k] = bias[nt];
#pragma unroll
      for (int kc = 0; kc < IK; ++kc)
        acc = __builtin_amdgcn_wmma_f32_16x16x32_f16(false, ax[kc].h16, false, fwih[nt][kc].h16,
                                                     (short)0, acc, false, false);
#pragma unroll
      for (int kc = 0; kc < 2; ++kc)
        acc = __builtin_amdgcn_wmma_f32_16x16x32_f16(false, ah[kc].h16, false, fwhh[nt][kc].h16,
                                                     (short)0, acc, false, false);
      accs[nt] = acc;
    }

    // ---- software-pipelined reload of this parity's x fragment (t+2) ----
    {
      const int tn = (t + 2 < T) ? (t + 2) : (T - 1);
      loadA(ax, tn);  // latency hidden behind activation + barrier + update
    }

    // ---- activation (branch-free sigmoid/tanh) + LDS gate store ----
#pragma unroll
    for (int nt = 0; nt < 4; ++nt) {
#pragma unroll
      for (int k = 0; k < 8; ++k) {
        const int m = k + (lane >> 4) * 8;  // C-layout row
        float a = accs[nt][k];
        float s = isg ? (a + a) : a;                        // tanh: use 2x
        float e = __builtin_amdgcn_exp2f(-1.442695041f * s);
        float r = __builtin_amdgcn_rcpf(1.0f + e);          // sigmoid(s)
        a = isg ? (r + r - 1.0f) : r;                       // tanh = 2*sig(2x)-1
        gbuf[wv][m][nt * 16 + lrow] = a;
      }
    }
    __syncthreads();

    // ---- cell/hidden update (128 threads x 8 elements) ----
    float hvals[8];
    v8h hv;
#pragma unroll
    for (int k = 0; k < 8; ++k) {
      const int n = un0 + k;
      const float ig = gbuf[0][um][n];
      const float fg = gbuf[1][um][n];
      const float gg = gbuf[2][um][n];
      const float og = gbuf[3][um][n];
      cst[k] = fg * cst[k] + ig * gg;
      hvals[k] = og * fast_tanh(cst[k]);
      hv[k] = (_Float16)hvals[k];
    }
    *reinterpret_cast<v8h*>(&hbuf[um][un0]) = hv;
    if (seq_out)
      *reinterpret_cast<v8h*>(seq_out + (((size_t)(b0 + um)) * T + t) * HH + un0) = hv;
    if (hT_out && t == T - 1) {
#pragma unroll
      for (int k = 0; k < 8; ++k)
        hT_out[(size_t)(b0 + um) * HH + un0 + k] = hvals[k];
    }
    __syncthreads();
  };

  for (int t = 0; t < T; t += 2) {  // T is even
    step(t, axA);
    step(t + 1, axB);
  }
}

// ---------------------------------------------------------------------------
// FC(64->128) + ReLU + FC(128->128) + LayerNorm. One block per batch row.
// ---------------------------------------------------------------------------
__global__ void __launch_bounds__(128)
head_kernel(const float* __restrict__ hT,    // [B,64]
            const float* __restrict__ fc1w,  // [128,64]
            const float* __restrict__ fc1b,  // [128]
            const float* __restrict__ fc2w,  // [128,128]
            const float* __restrict__ fc2b,  // [128]
            const float* __restrict__ lng,   // [128]
            const float* __restrict__ lnb,   // [128]
            float* __restrict__ out) {       // [B,128]
  __shared__ float hs[64];
  __shared__ float y1[128];
  __shared__ float red[128];
  const int b = blockIdx.x;
  const int j = threadIdx.x;
  if (j < 64) hs[j] = hT[(size_t)b * 64 + j];
  __syncthreads();

  float a = fc1b[j];
  const float* w1 = fc1w + (size_t)j * 64;
#pragma unroll 8
  for (int i = 0; i < 64; ++i) a += hs[i] * w1[i];
  y1[j] = fmaxf(a, 0.f);
  __syncthreads();

  float a2 = fc2b[j];
  const float* w2 = fc2w + (size_t)j * 128;
#pragma unroll 8
  for (int i = 0; i < 128; ++i) a2 += y1[i] * w2[i];

  red[j] = a2;
  __syncthreads();
  for (int s = 64; s > 0; s >>= 1) {
    if (j < s) red[j] += red[j + s];
    __syncthreads();
  }
  const float mu = red[0] / 128.f;
  __syncthreads();
  const float d = a2 - mu;
  red[j] = d * d;
  __syncthreads();
  for (int s = 64; s > 0; s >>= 1) {
    if (j < s) red[j] += red[j + s];
    __syncthreads();
  }
  const float var = red[0] / 128.f;
  out[(size_t)b * 128 + j] = d * rsqrtf(var + 1e-5f) * lng[j] + lnb[j];
}

// ---------------------------------------------------------------------------
extern "C" void kernel_launch(void* const* d_in, const int* in_sizes, int n_in,
                              void* d_out, int out_size, void* d_ws, size_t ws_size,
                              hipStream_t stream) {
  (void)in_sizes; (void)n_in; (void)out_size; (void)ws_size;
  const float* x    = (const float*)d_in[0];
  const float* wih1 = (const float*)d_in[1];
  const float* whh1 = (const float*)d_in[2];
  const float* bih1 = (const float*)d_in[3];
  const float* bhh1 = (const float*)d_in[4];
  const float* wih2 = (const float*)d_in[5];
  const float* whh2 = (const float*)d_in[6];
  const float* bih2 = (const float*)d_in[7];
  const float* bhh2 = (const float*)d_in[8];
  const float* fc1w = (const float*)d_in[9];
  const float* fc1b = (const float*)d_in[10];
  const float* fc2w = (const float*)d_in[11];
  const float* fc2b = (const float*)d_in[12];
  const float* lng  = (const float*)d_in[13];
  const float* lnb  = (const float*)d_in[14];
  float* out = (float*)d_out;

  const int B = 512, T = 1024, I1 = 32;

  char* ws = (char*)d_ws;
  _Float16* xh = (_Float16*)ws;                                           // B*T*32 f16 = 32 MB
  _Float16* h1 = (_Float16*)(ws + (size_t)B * T * I1 * sizeof(_Float16)); // B*T*64 f16 = 64 MB
  float*    hT = (float*)(ws + (size_t)B * T * I1 * sizeof(_Float16)
                             + (size_t)B * T * HH * sizeof(_Float16));    // B*64 f32

  {
    const int n4 = (B * T * I1) / 4;
    cvt_f32_f16<<<(n4 + 255) / 256, 256, 0, stream>>>(x, xh, n4);
  }
  lstm_layer_kernel<1><<<B / 16, 128, 0, stream>>>(xh, wih1, whh1, bih1, bhh1, h1, nullptr, T);
  lstm_layer_kernel<2><<<B / 16, 128, 0, stream>>>(h1, wih2, whh2, bih2, bhh2, nullptr, hT, T);
  head_kernel<<<B, 128, 0, stream>>>(hT, fc1w, fc1b, fc2w, fc2b, lng, lnb, out);
}
